// GraphNet_14285061226552
// MI455X (gfx1250) — matrix operand order repaired and checked by reading, not compiled, not run on hardware
//
#include <hip/hip_runtime.h>
#include <hip/hip_bf16.h>

// ---------------------------------------------------------------------------
// GIN 2-layer forward for MI455X (gfx1250, wave32, WMMA).
//   dims: 50000 nodes (= 3125 * 16, no tail), 800000 edges, 64->128->128->64
// Edge scatter via global_atomic_add_f32 (features L2-resident);
// MLPs via v_wmma_f32_16x16x32_bf16 with both linears fused per 16-row tile.
// v2: A operands hoisted into registers (loaded once per tile, not per
//     column tile), packed weights staged into LDS once per block and read
//     via ds_load by all 8 waves.
// ---------------------------------------------------------------------------

typedef __attribute__((ext_vector_type(16))) __bf16 v16bf;
typedef __attribute__((ext_vector_type(8)))  __bf16 v8bf;
typedef __attribute__((ext_vector_type(8)))  float  v8f;

#define NN_NODES 50000
#define NN_EDGES 800000
#define D_IN  64
#define D_HID 128
#define D_OUT 64
#define N_TILES (NN_NODES / 16)   // 3125

// ---------------------------------------------------------------------------
// Repack a row-major fp32 weight [K][N] into bf16 tiles matching the
// V_WMMA_*_BF16 B-operand layout: for tile (kc, nt), lane L holds the 16
// bf16 values B[kc*32 + (j>>3)*16 + (L>>4)*8 + (j&7)][nt*16 + (L&15)], j = 0..15,
// stored contiguously so the GEMM does one 32B load per lane.
// ---------------------------------------------------------------------------
__global__ void repack_weight_kernel(const float* __restrict__ W,
                                     __bf16* __restrict__ P, int K, int N) {
    int tid = blockIdx.x * blockDim.x + threadIdx.x;
    if (tid >= K * N) return;
    int NT      = N >> 4;
    int tileIdx = tid >> 9;        // / 512 (32 lanes * 16 elems)
    int within  = tid & 511;
    int lane    = within >> 4;
    int j       = within & 15;
    int kc = tileIdx / NT;
    int nt = tileIdx % NT;
    int k  = kc * 32 + ((j >> 3) << 4) + ((lane >> 4) << 3) + (j & 7);
    int n  = nt * 16 + (lane & 15);
    P[tid] = (__bf16)W[(size_t)k * N + n];
}

// ---------------------------------------------------------------------------
// Plain vectorized copy (fp32, count in float4 units).
// ---------------------------------------------------------------------------
__global__ void copy_f4_kernel(const float4* __restrict__ s,
                               float4* __restrict__ d, int n4) {
    int i = blockIdx.x * blockDim.x + threadIdx.x;
    if (i < n4) d[i] = s[i];
}

// ---------------------------------------------------------------------------
// Edge scatter-add: accum[dst[e]][:] += feat[src[e]][:].
// One thread per (edge, 4-float chunk); features stay hot in the 192MB L2.
// ---------------------------------------------------------------------------
template <int D>
__global__ void scatter_add_kernel(const float* __restrict__ feat,
                                   const long long* __restrict__ srcIdx,
                                   const long long* __restrict__ dstIdx,
                                   float* __restrict__ accum, int nEdges) {
    constexpr int CH = D / 4;
    int t = blockIdx.x * blockDim.x + threadIdx.x;
    int e = t / CH;
    int c = t % CH;
    if (e >= nEdges) return;
    long long s = srcIdx[e];
    long long d = dstIdx[e];
    const float4 v = *(const float4*)(feat + (size_t)s * D + c * 4);
    float* p = accum + (size_t)d * D + c * 4;
    atomicAdd(p + 0, v.x);
    atomicAdd(p + 1, v.y);
    atomicAdd(p + 2, v.z);
    atomicAdd(p + 3, v.w);
}

// ---------------------------------------------------------------------------
// Fused 2-layer MLP: hout = relu( relu(hin @ Wa + ba) @ Wb + bb )
// Hidden dim fixed at 128. One wave per 16-row node tile; the 16x128 bf16
// intermediate lives in a wave-private LDS region (in-order DS, no barrier).
// Weights live in LDS (staged once per block); A operands are register-
// resident across all output-column tiles.
//
// A layout (16-bit, 16x32): lane l<16 = row l, elems 0..7 -> K 0..7,
//   elems 8..15 -> K 16..23; lane l+16 = row l, K 8..15 / 24..31.
// C/D layout (f32 16x16): lane = column (lane&15), acc[i] = row (lane>>4)*8+i.
// ---------------------------------------------------------------------------
template <int DIN, int DOUT>
__global__ __launch_bounds__(256) void gin_mlp_kernel(
    const float* __restrict__ hin,
    const __bf16* __restrict__ WaP, const float* __restrict__ ba,
    const __bf16* __restrict__ WbP, const float* __restrict__ bb,
    float* __restrict__ hout, int nTiles) {
    constexpr int DH  = 128;
    constexpr int KC1 = DIN / 32;   // stage-1 K chunks
    constexpr int NT1 = DH / 16;    // stage-1 output column tiles
    constexpr int KC2 = DH / 32;
    constexpr int NT2 = DOUT / 16;
    constexpr int WA_ELEMS = DIN * DH;
    constexpr int WB_ELEMS = DH * DOUT;

    __shared__ __align__(16) __bf16 sWa[WA_ELEMS];
    __shared__ __align__(16) __bf16 sWb[WB_ELEMS];
    __shared__ __align__(16) __bf16 mid[8][16 * DH];  // 4KB per wave

    // --- Stage weights into LDS once per block (all 256 threads, 16B ea) ---
    {
        const int t = threadIdx.x;
#pragma unroll 2
        for (int i = t * 8; i < WA_ELEMS; i += 256 * 8)
            *(v8bf*)(sWa + i) = *(const v8bf*)(WaP + i);
#pragma unroll 2
        for (int i = t * 8; i < WB_ELEMS; i += 256 * 8)
            *(v8bf*)(sWb + i) = *(const v8bf*)(WbP + i);
    }
    __syncthreads();

    const int lane = threadIdx.x & 31;
    const int wave = threadIdx.x >> 5;
    int tile = blockIdx.x * 8 + wave;
    // Clamp instead of early-return: surplus waves redundantly recompute the
    // last tile (identical values, benign race) so EXEC stays all-ones for
    // WMMA and barrier participation stays uniform.
    if (tile >= nTiles) tile = nTiles - 1;

    const int half  = lane >> 4;   // which K half this lane holds
    const int nl    = lane & 15;   // column (B/C/D) or row (A) within tile
    const int mbase = tile * 16;

    __bf16* lmid = &mid[wave][0];

    // ---- Load the 16xDIN activation tile once into registers (A layout) ---
    v16bf aA[KC1];
#pragma unroll
    for (int kc = 0; kc < KC1; ++kc) {
        const float* rp = hin + (size_t)(mbase + nl) * DIN + kc * 32;
        const float4 c0 = *(const float4*)(rp + half * 8);
        const float4 c1 = *(const float4*)(rp + half * 8 + 4);
        const float4 c2 = *(const float4*)(rp + 16 + half * 8);
        const float4 c3 = *(const float4*)(rp + 16 + half * 8 + 4);
        v16bf a;
        a[0]  = (__bf16)c0.x; a[1]  = (__bf16)c0.y;
        a[2]  = (__bf16)c0.z; a[3]  = (__bf16)c0.w;
        a[4]  = (__bf16)c1.x; a[5]  = (__bf16)c1.y;
        a[6]  = (__bf16)c1.z; a[7]  = (__bf16)c1.w;
        a[8]  = (__bf16)c2.x; a[9]  = (__bf16)c2.y;
        a[10] = (__bf16)c2.z; a[11] = (__bf16)c2.w;
        a[12] = (__bf16)c3.x; a[13] = (__bf16)c3.y;
        a[14] = (__bf16)c3.z; a[15] = (__bf16)c3.w;
        aA[kc] = a;
    }

    // ---- Stage 1: lmid = relu(hin_tile @ Wa + ba)  (bf16 into LDS) ----
    for (int nt = 0; nt < NT1; ++nt) {
        v8f acc = {};
#pragma unroll
        for (int kc = 0; kc < KC1; ++kc) {
            const v16bf b =
                *(const v16bf*)(sWa + ((kc * NT1 + nt) * 32 + lane) * 16);
            acc = __builtin_amdgcn_wmma_f32_16x16x32_bf16(
                false, aA[kc], false, b, (short)0, acc, false, false);
        }
        const float bias = ba[nt * 16 + nl];
#pragma unroll
        for (int i = 0; i < 8; ++i) {
            float v = acc[i] + bias;
            v = v > 0.0f ? v : 0.0f;
            lmid[(half * 8 + i) * DH + nt * 16 + nl] = (__bf16)v;
        }
    }

    // ---- Load the 16x128 intermediate once into registers (A layout) ----
    // Wave-private LDS region: same-wave DS ops are in order; the compiler
    // inserts the s_wait_dscnt for the RAW dependency.
    v16bf aB[KC2];
#pragma unroll
    for (int kc = 0; kc < KC2; ++kc) {
        const __bf16* rp = lmid + nl * DH + kc * 32;
        const v8bf a0 = *(const v8bf*)(rp + half * 8);
        const v8bf a1 = *(const v8bf*)(rp + 16 + half * 8);
        v16bf a;
#pragma unroll
        for (int i = 0; i < 8; ++i) { a[i] = a0[i]; a[8 + i] = a1[i]; }
        aB[kc] = a;
    }

    // ---- Stage 2: hout_tile = relu(aB @ Wb + bb) ----
    for (int nt = 0; nt < NT2; ++nt) {
        v8f acc = {};
#pragma unroll
        for (int kc = 0; kc < KC2; ++kc) {
            const v16bf b =
                *(const v16bf*)(sWb + ((kc * NT2 + nt) * 32 + lane) * 16);
            acc = __builtin_amdgcn_wmma_f32_16x16x32_bf16(
                false, aB[kc], false, b, (short)0, acc, false, false);
        }
        const float bias = bb[nt * 16 + nl];
        float* op = hout + (size_t)mbase * DOUT + nt * 16 + nl;
#pragma unroll
        for (int i = 0; i < 8; ++i) {
            float v = acc[i] + bias;
            v = v > 0.0f ? v : 0.0f;
            op[(size_t)(half * 8 + i) * DOUT] = v;
        }
    }
}

// ---------------------------------------------------------------------------
// Host-side orchestration (graph-capture safe: only kernel launches).
// Inputs: x, edge_index, W1a, b1a, W1b, b1b, W2a, b2a, W2b, b2b
// ---------------------------------------------------------------------------
extern "C" void kernel_launch(void* const* d_in, const int* in_sizes, int n_in,
                              void* d_out, int out_size, void* d_ws,
                              size_t ws_size, hipStream_t stream) {
    (void)in_sizes; (void)n_in; (void)out_size; (void)ws_size;

    const float*     x    = (const float*)d_in[0];
    const long long* eidx = (const long long*)d_in[1];   // int64 [2, E]
    const float* W1a = (const float*)d_in[2];
    const float* b1a = (const float*)d_in[3];
    const float* W1b = (const float*)d_in[4];
    const float* b1b = (const float*)d_in[5];
    const float* W2a = (const float*)d_in[6];
    const float* b2a = (const float*)d_in[7];
    const float* W2b = (const float*)d_in[8];
    const float* b2b = (const float*)d_in[9];

    const long long* srcIdx = eidx;
    const long long* dstIdx = eidx + NN_EDGES;

    // Workspace carve-up (256B aligned regions).
    char*  ws  = (char*)d_ws;
    size_t off = 0;
    auto carve = [&](size_t bytes) -> void* {
        off = (off + 255) & ~(size_t)255;
        void* p = ws + off;
        off += bytes;
        return p;
    };
    float*  h0   = (float*)carve((size_t)NN_NODES * D_IN  * sizeof(float));
    float*  h1   = (float*)carve((size_t)NN_NODES * D_HID * sizeof(float));
    float*  h1b  = (float*)carve((size_t)NN_NODES * D_HID * sizeof(float));
    __bf16* W1aP = (__bf16*)carve((size_t)D_IN  * D_HID * sizeof(__bf16));
    __bf16* W1bP = (__bf16*)carve((size_t)D_HID * D_HID * sizeof(__bf16));
    __bf16* W2aP = (__bf16*)carve((size_t)D_HID * D_HID * sizeof(__bf16));
    __bf16* W2bP = (__bf16*)carve((size_t)D_HID * D_OUT * sizeof(__bf16));

    // 1) Repack weights into WMMA B-operand layout (tiny, once per launch).
    repack_weight_kernel<<<(D_IN * D_HID + 255) / 256, 256, 0, stream>>>(
        W1a, W1aP, D_IN, D_HID);
    repack_weight_kernel<<<(D_HID * D_HID + 255) / 256, 256, 0, stream>>>(
        W1b, W1bP, D_HID, D_HID);
    repack_weight_kernel<<<(D_HID * D_HID + 255) / 256, 256, 0, stream>>>(
        W2a, W2aP, D_HID, D_HID);
    repack_weight_kernel<<<(D_HID * D_OUT + 255) / 256, 256, 0, stream>>>(
        W2b, W2bP, D_HID, D_OUT);

    // 2) Layer 1 aggregation: h0 = x, then h0[dst] += x[src] over edges.
    {
        int n4 = NN_NODES * D_IN / 4;
        copy_f4_kernel<<<(n4 + 255) / 256, 256, 0, stream>>>(
            (const float4*)x, (float4*)h0, n4);
        int nth = NN_EDGES * (D_IN / 4);
        scatter_add_kernel<D_IN><<<(nth + 255) / 256, 256, 0, stream>>>(
            x, srcIdx, dstIdx, h0, NN_EDGES);
    }

    // 3) Layer 1 MLP (fused, WMMA): h1 = relu(relu(h0@W1a+b1a)@W1b+b1b).
    gin_mlp_kernel<D_IN, D_HID><<<(N_TILES + 7) / 8, 256, 0, stream>>>(
        h0, W1aP, b1a, W1bP, b1b, h1, N_TILES);

    // 4) Layer 2 aggregation: h1b = h1, then h1b[dst] += h1[src].
    {
        int n4 = NN_NODES * D_HID / 4;
        copy_f4_kernel<<<(n4 + 255) / 256, 256, 0, stream>>>(
            (const float4*)h1, (float4*)h1b, n4);
        int nth = NN_EDGES * (D_HID / 4);
        scatter_add_kernel<D_HID><<<(nth + 255) / 256, 256, 0, stream>>>(
            h1, srcIdx, dstIdx, h1b, NN_EDGES);
    }

    // 5) Layer 2 MLP (fused, WMMA) -> final output.
    gin_mlp_kernel<D_HID, D_OUT><<<(N_TILES + 7) / 8, 256, 0, stream>>>(
        h1b, W2aP, b2a, W2bP, b2b, (float*)d_out, N_TILES);
}
